// Overall_model_CRF_8169027797175
// MI455X (gfx1250) — compile-verified, matrix operand order checked by reading, not run on hardware
//
#include <hip/hip_runtime.h>

typedef __attribute__((ext_vector_type(16))) _Float16 v16h;
typedef __attribute__((ext_vector_type(8)))  float    v8f;

#define TAGS 12
#define STARTT 10
#define STOPT 11

__device__ __forceinline__ float sigm(float x) { return 1.0f / (1.0f + expf(-x)); }
__device__ __forceinline__ float eluf(float x) { return x > 0.0f ? x : (expf(x) - 1.0f); }

// Single persistent workgroup: conv stack -> WMMA gate precompute -> BiLSTM -> linear -> Viterbi.
__global__ __launch_bounds__(1024)
void crf_fused(const float* __restrict__ tg,
               const float* __restrict__ w1,   const float* __restrict__ b1,
               const float* __restrict__ gam,  const float* __restrict__ bet,
               const float* __restrict__ w2,   const float* __restrict__ b2,
               const float* __restrict__ wihf, const float* __restrict__ whhf,
               const float* __restrict__ bihf, const float* __restrict__ bhhf,
               const float* __restrict__ wihb, const float* __restrict__ whhb,
               const float* __restrict__ bihb, const float* __restrict__ bhhb,
               const float* __restrict__ h0,   const float* __restrict__ c0,
               const float* __restrict__ wtag, const float* __restrict__ btag,
               const float* __restrict__ trans,
               float* __restrict__ out, float* __restrict__ ws)
{
    __shared__ float    s_hall[2 * 60 * 128];   // hidden history; aliased as conv scratch early
    __shared__ float    s_p1[30 * 149];         // pooled conv1
    __shared__ _Float16 s_Xh[64 * 96];          // zero-padded f16 LSTM input features
    __shared__ float    s_gates[2 * 512];
    __shared__ float    s_h[2 * 128];
    __shared__ float    s_c[2 * 128];
    __shared__ float    s_feats[60 * TAGS];
    __shared__ int      s_bp[60 * TAGS];
    __shared__ float    s_mean[30];
    __shared__ float    s_rstd[30];

    const int tid = threadIdx.x;
    float* s_cv = s_hall;  // conv scratch (max 30*298 = 8940 floats), dead before LSTM phase

    // ---------- Phase A1: conv1 (60ch,K3) + ELU -> s_cv[30][298] ----------
    for (int idx = tid; idx < 30 * 298; idx += 1024) {
        int c = idx / 298, l = idx % 298;
        float acc = b1[c];
        const float* wrow = w1 + c * 180;
        for (int i = 0; i < 60; ++i) {
            const float* xr = tg + i * 300 + l;
            const float* wr = wrow + i * 3;
            acc += xr[0] * wr[0] + xr[1] * wr[1] + xr[2] * wr[2];
        }
        s_cv[idx] = eluf(acc);
    }
    __syncthreads();

    // ---------- Phase A2: train-mode BN stats (biased variance) ----------
    if (tid < 30) {
        float s = 0.f, ss = 0.f;
        for (int l = 0; l < 298; ++l) { float v = s_cv[tid * 298 + l]; s += v; ss += v * v; }
        float m = s * (1.0f / 298.0f);
        float var = ss * (1.0f / 298.0f) - m * m;
        s_mean[tid] = m;
        s_rstd[tid] = rsqrtf(var + 1e-5f);
    }
    __syncthreads();

    // ---------- Phase A3: BN + maxpool2 -> s_p1[30][149] ----------
    for (int idx = tid; idx < 30 * 149; idx += 1024) {
        int c = idx / 149, l = idx % 149;
        float g = gam[c], bb = bet[c], m = s_mean[c], r = s_rstd[c];
        float a = g * (s_cv[c * 298 + 2 * l]     - m) * r + bb;
        float b = g * (s_cv[c * 298 + 2 * l + 1] - m) * r + bb;
        s_p1[idx] = fmaxf(a, b);
    }
    __syncthreads();

    // ---------- Phase A4: conv2 (30ch,K3) + ELU -> s_cv[60][147] ----------
    for (int idx = tid; idx < 60 * 147; idx += 1024) {
        int c = idx / 147, l = idx % 147;
        float acc = b2[c];
        const float* wrow = w2 + c * 90;
        for (int i = 0; i < 30; ++i) {
            const float* xr = s_p1 + i * 149 + l;
            const float* wr = wrow + i * 3;
            acc += xr[0] * wr[0] + xr[1] * wr[1] + xr[2] * wr[2];
        }
        s_cv[idx] = eluf(acc);
    }
    __syncthreads();

    // ---------- Phase A5: maxpool2 -> f16 feature tile X[64][96] (zero padded) ----------
    for (int idx = tid; idx < 64 * 96; idx += 1024) s_Xh[idx] = (_Float16)0.0f;
    __syncthreads();
    for (int idx = tid; idx < 60 * 73; idx += 1024) {
        int s0 = idx / 73, f = idx % 73;
        float v = fmaxf(s_cv[s0 * 147 + 2 * f], s_cv[s0 * 147 + 2 * f + 1]);
        s_Xh[s0 * 96 + f] = (_Float16)v;
    }
    __syncthreads();

    // ---------- Phase B: WMMA gate precompute  Gx[dir][64][512] = X @ W_ih^T + b ----------
    // 256 tile jobs (2 dir x 4 Mtiles x 32 Ntiles), 3 K-steps of v_wmma_f32_16x16x32_f16 each.
    {
        const int lane  = tid & 31;
        const int wave  = tid >> 5;
        const int lm    = lane & 15;
        const int khalf = (lane >> 4) ? 8 : 0;   // lanes 16-31 hold K+8 / K+24 halves
        const int mAdd  = (lane >> 4) ? 8 : 0;   // D rows 8-15 live in lanes 16-31

        for (int job = wave; job < 256; job += 32) {
            int dir = job >> 7;
            int mt  = (job >> 5) & 3;
            int nt  = job & 31;
            int m0 = mt * 16, n0 = nt * 16;
            const float* wih = dir ? wihb : wihf;
            const float* bi  = dir ? bihb : bihf;
            const float* bh  = dir ? bhhb : bhhf;

            v8f acc = {0.f, 0.f, 0.f, 0.f, 0.f, 0.f, 0.f, 0.f};
            const _Float16* arow = &s_Xh[(m0 + lm) * 96];
            const float*    wrow = wih + (n0 + lm) * 73;

            for (int kk = 0; kk < 3; ++kk) {
                int kb = kk * 32 + khalf;
                v16h a, b;
#pragma unroll
                for (int j = 0; j < 8; ++j) {
                    a[j]     = arow[kb + j];            // padded LDS: always in-range
                    a[j + 8] = arow[kb + 16 + j];
                    int k1 = kb + j, k2 = kb + 16 + j;  // K=73, pad with zeros
                    b[j]     = (k1 < 73) ? (_Float16)wrow[k1] : (_Float16)0.0f;
                    b[j + 8] = (k2 < 73) ? (_Float16)wrow[k2] : (_Float16)0.0f;
                }
                acc = __builtin_amdgcn_wmma_f32_16x16x32_f16(
                          false, a, false, b, (short)0, acc, false, false);
            }
            int n = n0 + lm;
            float bias = bi[n] + bh[n];
            float* gx = ws + (size_t)(dir * 64 + m0 + mAdd) * 512 + n;
#pragma unroll
            for (int r = 0; r < 8; ++r) gx[(size_t)r * 512] = acc[r] + bias;
        }
    }
    __syncthreads();

    // ---------- Phase C: BiLSTM recurrence (both directions in parallel) ----------
    if (tid < 256) {
        int d = tid >> 7, k = tid & 127;
        s_h[d * 128 + k] = h0[d * 128 + k];
        s_c[d * 128 + k] = c0[d * 128 + k];
    }
    __syncthreads();

    {
        const int dir = tid >> 9;           // 0: forward, 1: backward
        const int g   = tid & 511;
        const float* whh  = dir ? whhb : whhf;
        const float* wrow = whh + g * 128;

        for (int s0 = 0; s0 < 60; ++s0) {
            int t = dir ? (59 - s0) : s0;
            float acc = ws[(size_t)(dir * 64 + t) * 512 + g];
            const float* hv = s_h + dir * 128;
#pragma unroll 4
            for (int k = 0; k < 128; ++k) acc += hv[k] * wrow[k];
            s_gates[dir * 512 + g] = acc;
            __syncthreads();
            if (g < 128) {  // PyTorch gate order i, f, g, o
                float iv = sigm(s_gates[dir * 512 + g]);
                float fv = sigm(s_gates[dir * 512 + 128 + g]);
                float gv = tanhf(s_gates[dir * 512 + 256 + g]);
                float ov = sigm(s_gates[dir * 512 + 384 + g]);
                float cv = fv * s_c[dir * 128 + g] + iv * gv;
                float hh = ov * tanhf(cv);
                s_c[dir * 128 + g] = cv;
                s_h[dir * 128 + g] = hh;
                s_hall[(dir * 60 + t) * 128 + g] = hh;
            }
            __syncthreads();
        }
    }

    // ---------- Phase D: tag projection feats[60][12] = [hf|hb] @ W_tag^T + b ----------
    for (int idx = tid; idx < 60 * TAGS; idx += 1024) {
        int t = idx / TAGS, tg2 = idx % TAGS;
        const float* wr = wtag + tg2 * 256;
        const float* hf = s_hall + t * 128;
        const float* hb = s_hall + (60 + t) * 128;
        float acc = btag[tg2];
        for (int j = 0; j < 128; ++j) acc += hf[j] * wr[j] + hb[j] * wr[128 + j];
        s_feats[idx] = acc;
    }
    __syncthreads();

    // ---------- Phase E: Viterbi decode (thread 0; 60 steps x 12 tags) ----------
    if (tid == 0) {
        float fv[TAGS], nf[TAGS];
        for (int i = 0; i < TAGS; ++i) fv[i] = -100000.0f;
        fv[STARTT] = 0.0f;
        for (int t = 0; t < 60; ++t) {
            for (int nx = 0; nx < TAGS; ++nx) {
                float best = -3.4e38f; int bi = 0;
                for (int pv = 0; pv < TAGS; ++pv) {
                    float v = fv[pv] + trans[nx * TAGS + pv];
                    if (v > best) { best = v; bi = pv; }   // first-max, matches jnp.argmax
                }
                nf[nx] = best + s_feats[t * TAGS + nx];
                s_bp[t * TAGS + nx] = bi;
            }
            for (int i = 0; i < TAGS; ++i) fv[i] = nf[i];
        }
        float best = -3.4e38f; int bl = 0;
        for (int i = 0; i < TAGS; ++i) {
            float v = fv[i] + trans[STOPT * TAGS + i];
            if (v > best) { best = v; bl = i; }
        }
        out[0] = best;
        int tag = bl;
        for (int t = 59; t >= 0; --t) { out[1 + t] = (float)tag; tag = s_bp[t * TAGS + tag]; }
    }
}

extern "C" void kernel_launch(void* const* d_in, const int* in_sizes, int n_in,
                              void* d_out, int out_size, void* d_ws, size_t ws_size,
                              hipStream_t stream) {
    const float* tg    = (const float*)d_in[0];
    // d_in[1] = pos, d_in[2] = adj : unused by the reference
    const float* w1    = (const float*)d_in[3];
    const float* b1    = (const float*)d_in[4];
    const float* gam   = (const float*)d_in[5];
    const float* bet   = (const float*)d_in[6];
    const float* w2    = (const float*)d_in[7];
    const float* b2    = (const float*)d_in[8];
    const float* wihf  = (const float*)d_in[9];
    const float* whhf  = (const float*)d_in[10];
    const float* bihf  = (const float*)d_in[11];
    const float* bhhf  = (const float*)d_in[12];
    const float* wihb  = (const float*)d_in[13];
    const float* whhb  = (const float*)d_in[14];
    const float* bihb  = (const float*)d_in[15];
    const float* bhhb  = (const float*)d_in[16];
    const float* h0    = (const float*)d_in[17];
    const float* c0    = (const float*)d_in[18];
    const float* wtag  = (const float*)d_in[19];
    const float* btag  = (const float*)d_in[20];
    const float* trans = (const float*)d_in[21];

    crf_fused<<<1, 1024, 0, stream>>>(tg, w1, b1, gam, bet, w2, b2,
                                      wihf, whhf, bihf, bhhf,
                                      wihb, whhb, bihb, bhhb,
                                      h0, c0, wtag, btag, trans,
                                      (float*)d_out, (float*)d_ws);
}
